// LSTM_54760833024025
// MI455X (gfx1250) — matrix operand order repaired and checked by reading, not compiled, not run on hardware
//
#include <hip/hip_runtime.h>
#include <hip/hip_bf16.h>
#include <math.h>

// Problem constants (from reference): H=1024, B=256, T=256, C=10
constexpr int kH  = 1024;   // hidden
constexpr int kB  = 256;    // batch
constexpr int kT  = 256;    // timesteps
constexpr int kC  = 10;     // classes

typedef __attribute__((ext_vector_type(16))) __bf16       bf16x16;
typedef __attribute__((ext_vector_type(8)))  float        f32x8;
typedef __attribute__((ext_vector_type(4)))  unsigned int u32x4;

union FragU  { u32x4 q[2]; bf16x16 v; };
union HPack  { u32x4 q;    __bf16 h[8]; };

// ---------------------------------------------------------------------------
// Pack W_gh/W_ih/W_fh/W_oh (fp32, each 1024x1024 row-major) into one bf16
// W_all (4096x1024 row-major): rows [0,1024)=g, [1024,2048)=i, ...
// ---------------------------------------------------------------------------
__global__ void lstm_pack_w(const float* __restrict__ Wgh, const float* __restrict__ Wih,
                            const float* __restrict__ Wfh, const float* __restrict__ Woh,
                            __bf16* __restrict__ Wall) {
    int idx = blockIdx.x * 256 + threadIdx.x;      // 0 .. 4096*1024-1
    int r = idx >> 10;
    int k = idx & (kH - 1);
    const float* src = (r < kH) ? Wgh : (r < 2 * kH) ? Wih : (r < 3 * kH) ? Wfh : Woh;
    int rr = r & (kH - 1);
    Wall[idx] = (__bf16)src[rr * kH + k];
}

// Zero-init cell state + hidden state buffers (workspace is poisoned 0xAA).
__global__ void lstm_init(float* __restrict__ c, float* __restrict__ hf,
                          __bf16* __restrict__ hbf) {
    int idx = blockIdx.x * 256 + threadIdx.x;      // 0 .. H*B-1
    c[idx]   = 0.0f;
    hf[idx]  = 0.0f;
    hbf[idx] = (__bf16)0.0f;
}

// ---------------------------------------------------------------------------
// Fully fused LSTM step. Each block owns hidden rows [32*bx, 32*bx+32) for
// ALL FOUR gates and batch cols [64*by, 64*by+64).
//   - 8 waves: wave&3 = gate (g,i,f,o), wave>>2 = N half (2 x 32 cols).
//   - Wave tile 32x32 = 2x2 fragments of v_wmma_f32_16x16x32_bf16 over K=1024.
//   - Gate tiles exchanged via LDS (padded rows, no bank conflicts), then the
//     cell update (tanh/sigmoid, c/h) runs in the epilogue: gate
//     pre-activations never touch global memory.
// h lives transposed as hbf (B x H, bf16) so the B operand's K is contiguous.
// Grid: (1024/32, 256/64) = (32, 4), 256 threads.
// ---------------------------------------------------------------------------
__global__ __launch_bounds__(256)
void lstm_step_fused(const __bf16* __restrict__ Wall, const float* __restrict__ x,
                     const float* __restrict__ Wgx, const float* __restrict__ Wix,
                     const float* __restrict__ Wfx, const float* __restrict__ Wox,
                     const float* __restrict__ bg,  const float* __restrict__ bi,
                     const float* __restrict__ bfv, const float* __restrict__ bo,
                     float* __restrict__ c, float* __restrict__ hf,
                     __bf16* __restrict__ hbf, int t) {
    constexpr int LPAD = 65;                   // 64 cols + 1 pad: conflict-free
    __shared__ float sg[4][32][LPAD];          // 4 gate tiles, 32 rows x 64 cols

    const int tid  = threadIdx.x;
    const int wave = tid >> 5;
    const int lane = tid & 31;
    const int half = lane >> 4;                // 0: lanes 0-15, 1: lanes 16-31
    const int l16  = lane & 15;

    const int gate  = wave & 3;                // one gate per M-wave
    const int waveN = wave >> 2;               // 2 waves across N
    const int hrowB = blockIdx.x * 32;         // hidden-row base of this block
    const int colB  = blockIdx.y * 64;         // batch-col base of this block
    const int colW  = colB + waveN * 32;       // this wave's col base

    // A rows in Wall: gate-major fused weight matrix (4096 x 1024).
    const __bf16* aRow[2];
    const __bf16* bCol[2];
    aRow[0] = Wall + (size_t)(gate * kH + hrowB + l16)      * kH;
    aRow[1] = Wall + (size_t)(gate * kH + hrowB + 16 + l16) * kH;
    bCol[0] = hbf  + (size_t)(colW + l16)      * kH;
    bCol[1] = hbf  + (size_t)(colW + 16 + l16) * kH;

    f32x8 acc[2][2] = {};

    for (int k0 = 0; k0 < kH; k0 += 32) {
        FragU a[2], b[2];
        // A fragment (16x32): v0..3 <- K = k0+8*half.., v4..7 <- K = k0+16+8*half..
#pragma unroll
        for (int mi = 0; mi < 2; ++mi) {
            a[mi].q[0] = *(const u32x4*)(aRow[mi] + k0 + 8 * half);
            a[mi].q[1] = *(const u32x4*)(aRow[mi] + k0 + 16 + 8 * half);
        }
        // B fragment (32x16): v0..3 <- K = k0+8*half.., v4..7 <- K = k0+16+8*half..
#pragma unroll
        for (int ni = 0; ni < 2; ++ni) {
            b[ni].q[0] = *(const u32x4*)(bCol[ni] + k0 + 8 * half);
            b[ni].q[1] = *(const u32x4*)(bCol[ni] + k0 + 16 + 8 * half);
        }
#pragma unroll
        for (int mi = 0; mi < 2; ++mi)
#pragma unroll
            for (int ni = 0; ni < 2; ++ni)
                acc[mi][ni] = __builtin_amdgcn_wmma_f32_16x16x32_bf16(
                    false, a[mi].v, false, b[ni].v,
                    (short)0, acc[mi][ni], false, false);
    }

    // Scatter gate pre-activations into LDS (C/D layout: M = 8*half + r, N = l16).
#pragma unroll
    for (int mi = 0; mi < 2; ++mi)
#pragma unroll
        for (int ni = 0; ni < 2; ++ni) {
            const int lc = waveN * 32 + ni * 16 + l16;
#pragma unroll
            for (int r = 0; r < 8; ++r) {
                const int lr = mi * 16 + half * 8 + r;
                sg[gate][lr][lc] = acc[mi][ni][r];
            }
        }
    __syncthreads();

    // Epilogue: cell update for 32 rows x 64 cols = 8 elems/thread.
    // thread -> col = tid&63, row group = tid>>6 (8 consecutive rows).
    const int col = tid & 63;
    const int rg  = tid >> 6;
    const int b   = colB + col;
    const float xb = x[b * kT + t];            // x is (B, T) row-major
    const float bgv = bg[b], biv = bi[b], bfvv = bfv[b], bov = bo[b];

    HPack hp;
#pragma unroll
    for (int r8 = 0; r8 < 8; ++r8) {
        const int lr   = rg * 8 + r8;
        const int hrow = hrowB + lr;
        const int idx  = hrow * kB + b;        // (H, B) row-major

        float zg = sg[0][lr][col] + Wgx[hrow] * xb + bgv;
        float zi = sg[1][lr][col] + Wix[hrow] * xb + biv;
        float zf = sg[2][lr][col] + Wfx[hrow] * xb + bfvv;
        float zo = sg[3][lr][col] + Wox[hrow] * xb + bov;

        float g  = tanhf(zg);
        float ii = 1.0f / (1.0f + expf(-zi));
        float ff = 1.0f / (1.0f + expf(-zf));
        float oo = 1.0f / (1.0f + expf(-zo));

        float cn = g * ii + c[idx] * ff;
        c[idx] = cn;
        float hn = tanhf(cn) * oo;

        hf[idx]  = hn;                         // fp32 (H,B) for final projection
        hp.h[r8] = (__bf16)hn;                 // bf16, 8 consecutive hidden rows
    }
    // One 16-byte store of 8 bf16 into hbf (B x H): offset multiple of 16 B.
    *(u32x4*)(hbf + (size_t)b * kH + hrowB + rg * 8) = hp.q;
}

// ---------------------------------------------------------------------------
// Final projection p = W_ph(10x1024) @ h(1024x256) + b_p (per-column), then
// softmax over axis=1 (the 256-wide batch axis, matching the reference),
// output y.T as (256, 10) row-major.
// ---------------------------------------------------------------------------
__global__ void lstm_proj_softmax(const float* __restrict__ Wph, const float* __restrict__ bp,
                                  const float* __restrict__ hf, float* __restrict__ out) {
    __shared__ float sp[kC][kB];
    __shared__ float smax[kC];
    __shared__ float ssum[kC];

    int b = threadIdx.x;                       // 0..255
    float acc[kC];
#pragma unroll
    for (int c = 0; c < kC; ++c) acc[c] = 0.0f;

    for (int k = 0; k < kH; ++k) {
        float hv = hf[k * kB + b];
#pragma unroll
        for (int c = 0; c < kC; ++c)
            acc[c] = fmaf(Wph[c * kH + k], hv, acc[c]);
    }
#pragma unroll
    for (int c = 0; c < kC; ++c) sp[c][b] = acc[c] + bp[b];
    __syncthreads();

    if (b < kC) {
        float m = -1e30f;
        for (int j = 0; j < kB; ++j) m = fmaxf(m, sp[b][j]);
        float s = 0.0f;
        for (int j = 0; j < kB; ++j) s += expf(sp[b][j] - m);
        smax[b] = m;
        ssum[b] = s;
    }
    __syncthreads();

#pragma unroll
    for (int c = 0; c < kC; ++c)
        out[b * kC + c] = expf(sp[c][b] - smax[c]) / ssum[c];
}

// ---------------------------------------------------------------------------
extern "C" void kernel_launch(void* const* d_in, const int* in_sizes, int n_in,
                              void* d_out, int out_size, void* d_ws, size_t ws_size,
                              hipStream_t stream) {
    (void)in_sizes; (void)n_in; (void)out_size; (void)ws_size;

    const float* x   = (const float*)d_in[0];
    const float* Wgx = (const float*)d_in[1];
    const float* Wgh = (const float*)d_in[2];
    const float* Wix = (const float*)d_in[3];
    const float* Wih = (const float*)d_in[4];
    const float* Wfx = (const float*)d_in[5];
    const float* Wfh = (const float*)d_in[6];
    const float* Wox = (const float*)d_in[7];
    const float* Woh = (const float*)d_in[8];
    const float* Wph = (const float*)d_in[9];
    const float* bg  = (const float*)d_in[10];
    const float* bi  = (const float*)d_in[11];
    const float* bf  = (const float*)d_in[12];
    const float* bo  = (const float*)d_in[13];
    const float* bp  = (const float*)d_in[14];

    // Workspace layout (bytes):
    //   [0,   8M)   Wall  bf16 4096x1024
    //   [8M,  9M)   c     fp32 1024x256
    //   [9M, 10M)   hf    fp32 1024x256
    //   [10M,10.5M) hbf   bf16 256x1024
    char* ws = (char*)d_ws;
    __bf16* Wall = (__bf16*)(ws);
    float*  cbuf = (float*) (ws + (8u  << 20));
    float*  hf   = (float*) (ws + (9u  << 20));
    __bf16* hbf  = (__bf16*)(ws + (10u << 20));

    // One-time per-call prep (deterministic, no caching).
    lstm_pack_w<<<(4 * kH * kH) / 256, 256, 0, stream>>>(Wgh, Wih, Wfh, Woh, Wall);
    lstm_init  <<<(kH * kB)     / 256, 256, 0, stream>>>(cbuf, hf, hbf);

    const dim3 stepGrid(kH / 32, kB / 64);     // (32, 4) = 128 blocks
    for (int t = 0; t < kT; ++t) {
        lstm_step_fused<<<stepGrid, 256, 0, stream>>>(
            Wall, x, Wgx, Wix, Wfx, Wox, bg, bi, bf, bo, cbuf, hf, hbf, t);
    }

    lstm_proj_softmax<<<1, kB, 0, stream>>>(Wph, bp, hf, (float*)d_out);
}
